// GRU_89489938579752
// MI455X (gfx1250) — matrix operand order repaired and checked by reading, not compiled, not run on hardware
//
#include <hip/hip_runtime.h>
#include <stdint.h>

#define T_SEQ 512
#define B_    256
#define H_    1024
#define E_    256
#define K_    1280   // H_ + E_
#define C_    128

typedef __attribute__((ext_vector_type(16))) __bf16 v16bf;
typedef __attribute__((ext_vector_type(8)))  float  v8f;

union BF16Vec {
    v16bf v;
    uint4 q[2];
};

__device__ __forceinline__ unsigned short f2bf(float f) {
    unsigned u = __float_as_uint(f);
    u += 0x7FFFu + ((u >> 16) & 1u);           // round-to-nearest-even
    return (unsigned short)(u >> 16);
}

__device__ __forceinline__ float sigmoidf_(float x) {
    return 1.0f / (1.0f + __expf(-x));
}

// ---------------------------------------------------------------------------
// Software-pipelined 16(M)x64(N) x K=1280 bf16 WMMA tile.
//   aH: A h-part base, already offset by lane (mA*H_ + kbA)
//   aX: A x-part base, already offset by lane (mA*E_ + kbA)
//   bB: B base, already offset by lane (nB*K_ + kbB); columns step 16*K_.
// Double-buffers A and B registers: loads for chunk k+32 are issued before
// the WMMAs of chunk k, so s_wait_loadcnt can be partial and L2 traffic
// overlaps matrix math.
// ---------------------------------------------------------------------------
__device__ __forceinline__ void gemm_tile_16x64(
    const unsigned short* __restrict__ aH,
    const unsigned short* __restrict__ aX,
    const unsigned short* __restrict__ bB,
    v8f acc[4])
{
    BF16Vec a0, a1;
    BF16Vec b00, b01, b02, b03;
    BF16Vec b10, b11, b12, b13;

    auto ldA = [&](BF16Vec& a, int kk) {
        const unsigned short* p = (kk < H_) ? (aH + kk) : (aX + (kk - H_));
        a.q[0] = *(const uint4*)(p);
        a.q[1] = *(const uint4*)(p + 16);
    };
    auto ldB = [&](BF16Vec& b, int j, int kk) {
        const unsigned short* p = bB + (size_t)j * (16 * K_) + kk;
        b.q[0] = *(const uint4*)(p);
        b.q[1] = *(const uint4*)(p + 16);
    };

    // Prologue: chunk 0 into buffer 0.
    ldA(a0, 0);
    ldB(b00, 0, 0); ldB(b01, 1, 0); ldB(b02, 2, 0); ldB(b03, 3, 0);

    for (int kk = 0; kk < K_; kk += 64) {
        // Prefetch chunk kk+32 into buffer 1 (always in range: kk<=1216).
        const int k1 = kk + 32;
        ldA(a1, k1);
        ldB(b10, 0, k1); ldB(b11, 1, k1); ldB(b12, 2, k1); ldB(b13, 3, k1);

        // Compute chunk kk from buffer 0.
        acc[0] = __builtin_amdgcn_wmma_f32_16x16x32_bf16(false, a0.v, false, b00.v,
                                                         (short)0, acc[0], false, false);
        acc[1] = __builtin_amdgcn_wmma_f32_16x16x32_bf16(false, a0.v, false, b01.v,
                                                         (short)0, acc[1], false, false);
        acc[2] = __builtin_amdgcn_wmma_f32_16x16x32_bf16(false, a0.v, false, b02.v,
                                                         (short)0, acc[2], false, false);
        acc[3] = __builtin_amdgcn_wmma_f32_16x16x32_bf16(false, a0.v, false, b03.v,
                                                         (short)0, acc[3], false, false);

        // Prefetch chunk kk+64 into buffer 0 (wraps to 0 on last iter; harmless).
        const int k2 = (kk + 64 < K_) ? (kk + 64) : 0;
        ldA(a0, k2);
        ldB(b00, 0, k2); ldB(b01, 1, k2); ldB(b02, 2, k2); ldB(b03, 3, k2);

        // Compute chunk kk+32 from buffer 1.
        acc[0] = __builtin_amdgcn_wmma_f32_16x16x32_bf16(false, a1.v, false, b10.v,
                                                         (short)0, acc[0], false, false);
        acc[1] = __builtin_amdgcn_wmma_f32_16x16x32_bf16(false, a1.v, false, b11.v,
                                                         (short)0, acc[1], false, false);
        acc[2] = __builtin_amdgcn_wmma_f32_16x16x32_bf16(false, a1.v, false, b12.v,
                                                         (short)0, acc[2], false, false);
        acc[3] = __builtin_amdgcn_wmma_f32_16x16x32_bf16(false, a1.v, false, b13.v,
                                                         (short)0, acc[3], false, false);
    }
}

// ---------------------------------------------------------------------------
// One-time: convert W_z, W_r, W (fp32, [K=1280][N=1024] row-major) into
// transposed bf16 [N=1024][K=1280] so WMMA B-operand K-slices are contiguous.
// ---------------------------------------------------------------------------
__global__ void __launch_bounds__(256) wconv_kernel(
    const float* __restrict__ Wz, const float* __restrict__ Wr,
    const float* __restrict__ Wh,
    unsigned short* __restrict__ Wzt, unsigned short* __restrict__ Wrt,
    unsigned short* __restrict__ Wht)
{
    const int per = K_ * H_;
    int tid = blockIdx.x * blockDim.x + threadIdx.x;
    int w = tid / per;
    if (w >= 3) return;
    int rem = tid - w * per;
    int k = rem / H_;
    int n = rem - k * H_;
    const float* src = (w == 0) ? Wz : (w == 1) ? Wr : Wh;
    unsigned short* dst = (w == 0) ? Wzt : (w == 1) ? Wrt : Wht;
    dst[(size_t)n * K_ + k] = f2bf(src[rem]);
}

// ---------------------------------------------------------------------------
// One-time: h = 0 (fp32 + bf16 mirrors), gather x_0 = emb[token[b,0]] as bf16
// ---------------------------------------------------------------------------
__global__ void __launch_bounds__(256) init_kernel(
    float* __restrict__ hfp, unsigned short* __restrict__ hbf,
    unsigned short* __restrict__ x0,
    const int* __restrict__ tok, const float* __restrict__ emb)
{
    int tid = blockIdx.x * blockDim.x + threadIdx.x;
    if (tid < B_ * H_) { hfp[tid] = 0.0f; hbf[tid] = 0; }
    if (tid < B_ * E_) {
        int b = tid >> 8;         // / E_
        int e = tid & (E_ - 1);
        int tk = tok[b * T_SEQ];  // t = 0
        x0[tid] = f2bf(emb[(size_t)tk * E_ + e]);
    }
}

// ---------------------------------------------------------------------------
// Per-step kernel 1: Z = sigmoid([h|x] @ Wz), R = sigmoid([h|x] @ Wr)
// writes zfp (fp32) and rhbf = bf16(R * h).
// Grid: 64 blocks (blockIdx>>5 selects z/r), 256 threads = 8 waves.
// Wave tile: 16(M) x 64(N); block tile: 32(M) x 256(N).
// ---------------------------------------------------------------------------
__global__ void __launch_bounds__(256) zr_kernel(
    const unsigned short* __restrict__ hbf,
    const unsigned short* __restrict__ xbf,
    const unsigned short* __restrict__ Wzt,
    const unsigned short* __restrict__ Wrt,
    const float* __restrict__ hfp,
    float* __restrict__ zfp,
    unsigned short* __restrict__ rhbf)
{
    const int lane = threadIdx.x & 31;
    const int wid  = threadIdx.x >> 5;
    const int w    = blockIdx.x >> 5;      // 0 = z, 1 = r
    const int blk  = blockIdx.x & 31;
    const int row0 = (blk & 7) * 32 + (wid & 1) * 16;
    const int col0 = (blk >> 3) * 256 + (wid >> 1) * 64;

    const unsigned short* __restrict__ Wt = (w == 0) ? Wzt : Wrt;

    // A-operand lane addressing (16-bit A 16x32 layout, ISA §7.12.2)
    const int mA  = row0 + (lane & 15);
    const int kbA = (lane >> 4) * 8;
    // B-operand lane addressing (same K order as A: 8-element runs at 8h and 16+8h)
    const int nB  = col0 + (lane & 15);
    const int kbB = (lane >> 4) * 8;

    v8f acc[4] = {};
    gemm_tile_16x64(hbf + (size_t)mA * H_ + kbA,
                    xbf + (size_t)mA * E_ + kbA,
                    Wt  + (size_t)nB * K_ + kbB,
                    acc);

    // C/D layout: VGPR v -> M = row0 + v + (lane>>4)*8, N = col0 + j*16 + (lane&15)
    const int rr = (lane >> 4) * 8;
    const int cc = lane & 15;
#pragma unroll
    for (int j = 0; j < 4; ++j) {
#pragma unroll
        for (int v = 0; v < 8; ++v) {
            int m = row0 + v + rr;
            int n = col0 + j * 16 + cc;
            size_t idx = (size_t)m * H_ + n;
            float s = sigmoidf_(acc[j][v]);
            if (w == 0) zfp[idx] = s;
            else        rhbf[idx] = f2bf(s * hfp[idx]);
        }
    }
}

// ---------------------------------------------------------------------------
// Per-step kernel 2: h_tilde = tanh([r*h|x] @ W); h = h + z*(h_tilde - h)
// Also gathers x_{t+1} = bf16(emb[token[b, t+1]]) into the other x buffer.
// Grid: 32 blocks, 256 threads.
// ---------------------------------------------------------------------------
__global__ void __launch_bounds__(256) h_kernel(
    const unsigned short* __restrict__ rhbf,
    const unsigned short* __restrict__ xcur,
    const unsigned short* __restrict__ Wht,
    float* __restrict__ hfp,
    const float* __restrict__ zfp,
    unsigned short* __restrict__ hbf,
    unsigned short* __restrict__ xnext,
    const int* __restrict__ tok,
    const float* __restrict__ emb,
    int t)
{
    const int lane = threadIdx.x & 31;
    const int wid  = threadIdx.x >> 5;
    const int blk  = blockIdx.x;
    const int row0 = (blk & 7) * 32 + (wid & 1) * 16;
    const int col0 = (blk >> 3) * 256 + (wid >> 1) * 64;

    const int mA  = row0 + (lane & 15);
    const int kbA = (lane >> 4) * 8;
    const int nB  = col0 + (lane & 15);
    const int kbB = (lane >> 4) * 8;

    v8f acc[4] = {};
    gemm_tile_16x64(rhbf + (size_t)mA * H_ + kbA,
                    xcur + (size_t)mA * E_ + kbA,
                    Wht  + (size_t)nB * K_ + kbB,
                    acc);

    const int rr = (lane >> 4) * 8;
    const int cc = lane & 15;
#pragma unroll
    for (int j = 0; j < 4; ++j) {
#pragma unroll
        for (int v = 0; v < 8; ++v) {
            int m = row0 + v + rr;
            int n = col0 + j * 16 + cc;
            size_t idx = (size_t)m * H_ + n;
            float ht = tanhf(acc[j][v]);
            float h  = hfp[idx];
            float z  = zfp[idx];
            float hn = fmaf(ht - h, z, h);   // h*(1-z) + ht*z
            hfp[idx] = hn;
            hbf[idx] = f2bf(hn);
        }
    }

    // Gather next timestep's embeddings into the other x buffer (no overlap
    // with xcur, so no intra-kernel race). 32 blocks * 256 thr * 8 = 65536.
    int tp1 = t + 1;
    if (tp1 < T_SEQ) {
        int g  = blockIdx.x * 256 + threadIdx.x;   // 0..8191
        int b  = g >> 5;
        int e0 = (g & 31) * 8;
        int tk = tok[b * T_SEQ + tp1];
        const float* es = emb + (size_t)tk * E_ + e0;
        unsigned short* xd = xnext + (size_t)b * E_ + e0;
#pragma unroll
        for (int i = 0; i < 8; ++i) xd[i] = f2bf(es[i]);
    }
}

// ---------------------------------------------------------------------------
// Final: p = h @ W_ph + b; out = log_softmax(p). One block per batch row.
// ---------------------------------------------------------------------------
__global__ void __launch_bounds__(128) head_kernel(
    const float* __restrict__ hfp,
    const float* __restrict__ Wph,
    const float* __restrict__ bp,
    float* __restrict__ out)
{
    __shared__ float s[C_];
    int b = blockIdx.x;
    int c = threadIdx.x;
    float acc = bp[c];
    const float* hrow = hfp + (size_t)b * H_;
    for (int k = 0; k < H_; ++k)
        acc = fmaf(hrow[k], Wph[(size_t)k * C_ + c], acc);

    s[c] = acc; __syncthreads();
    for (int off = 64; off > 0; off >>= 1) {
        if (c < off) s[c] = fmaxf(s[c], s[c + off]);
        __syncthreads();
    }
    float mx = s[0]; __syncthreads();
    float ex = __expf(acc - mx);
    s[c] = ex; __syncthreads();
    for (int off = 64; off > 0; off >>= 1) {
        if (c < off) s[c] = s[c] + s[c + off];
        __syncthreads();
    }
    float lse = logf(s[0]);
    out[(size_t)b * C_ + c] = acc - mx - lse;
}

// ---------------------------------------------------------------------------
// Workspace layout (bytes, 256-aligned):
//   Wzt 0          (2,621,440)   bf16 [1024][1280]
//   Wrt 2,621,440  (2,621,440)
//   Wht 5,242,880  (2,621,440)
//   hfp 7,864,320  (1,048,576)   fp32 [256][1024]
//   zfp 8,912,896  (1,048,576)
//   hbf 9,961,472  (  524,288)   bf16 [256][1024]
//   rhbf 10,485,760 ( 524,288)
//   xb0 11,010,048 (  131,072)   bf16 [256][256]
//   xb1 11,141,120 (  131,072)
// Total ~11.3 MB.
// ---------------------------------------------------------------------------
extern "C" void kernel_launch(void* const* d_in, const int* in_sizes, int n_in,
                              void* d_out, int out_size, void* d_ws, size_t ws_size,
                              hipStream_t stream) {
    const int*   x    = (const int*)d_in[0];
    const float* emb  = (const float*)d_in[1];
    const float* Wz   = (const float*)d_in[2];
    const float* Wr   = (const float*)d_in[3];
    const float* Wh   = (const float*)d_in[4];
    const float* Wph  = (const float*)d_in[5];
    const float* bp   = (const float*)d_in[6];
    float*       out  = (float*)d_out;

    char* ws = (char*)d_ws;
    unsigned short* Wzt  = (unsigned short*)(ws + 0);
    unsigned short* Wrt  = (unsigned short*)(ws + 2621440);
    unsigned short* Wht  = (unsigned short*)(ws + 5242880);
    float*          hfp  = (float*)(ws + 7864320);
    float*          zfp  = (float*)(ws + 8912896);
    unsigned short* hbf  = (unsigned short*)(ws + 9961472);
    unsigned short* rhbf = (unsigned short*)(ws + 10485760);
    unsigned short* xb0  = (unsigned short*)(ws + 11010048);
    unsigned short* xb1  = (unsigned short*)(ws + 11141120);

    wconv_kernel<<<15360, 256, 0, stream>>>(Wz, Wr, Wh, Wzt, Wrt, Wht);
    init_kernel<<<1024, 256, 0, stream>>>(hfp, hbf, xb0, x, emb);

    for (int t = 0; t < T_SEQ; ++t) {
        unsigned short* xc = (t & 1) ? xb1 : xb0;
        unsigned short* xn = (t & 1) ? xb0 : xb1;
        zr_kernel<<<64, 256, 0, stream>>>(hbf, xc, Wzt, Wrt, hfp, zfp, rhbf);
        h_kernel<<<32, 256, 0, stream>>>(rhbf, xc, Wht, hfp, zfp, hbf, xn, x, emb, t);
    }

    head_kernel<<<B_, C_, 0, stream>>>(hfp, Wph, bp, out);
}
